// GATModel_33406255628524
// MI455X (gfx1250) — matrix operand order repaired and checked by reading, not compiled, not run on hardware
//
#include <hip/hip_runtime.h>
#include <math.h>

// ---- problem constants (match reference) ----
#define NNODES   100000
#define IN_CH    128
#define OUT_CH   32
#define HEADS    4
#define NEDGES   1600000
#define EPRIME   (NEDGES + NNODES)   // edges + self loops
#define FDIM     (HEADS * OUT_CH)    // 128
#define NEG_SLOPE 0.2f

typedef __bf16 bf16;
typedef __attribute__((ext_vector_type(8)))  __bf16 v8bf;   // 16 bytes
typedef __attribute__((ext_vector_type(16))) __bf16 v16bf;  // 32 bytes
typedef __attribute__((ext_vector_type(8)))  float  v8f;

#define LDS_K (IN_CH + 8)   // 136 bf16 = 272 B row stride: keeps 16B alignment, skews banks

// ------------------------------------------------------------------
// 0) init: zero accumulator (d_out), amax = -inf, denom = 0
// ------------------------------------------------------------------
__global__ void gat_init(float* __restrict__ out, float* __restrict__ amax,
                         float* __restrict__ denom) {
  long long i = (long long)blockIdx.x * blockDim.x + threadIdx.x;
  if (i < (long long)NNODES * FDIM) out[i] = 0.0f;
  if (i < (long long)NNODES * HEADS) { amax[i] = -INFINITY; denom[i] = 0.0f; }
}

// ------------------------------------------------------------------
// 1) WMMA GEMM: XW[N,128] = X[N,128] @ W[128,128], bf16 in / f32 acc.
//    Block = 256 threads = 8 waves; block computes 64 rows x 128 cols.
//    Each wave owns one 16-wide N tile and iterates 4 M tiles; the 4
//    B fragments (one per K step) are loaded once and reused in regs.
// ------------------------------------------------------------------
__global__ __launch_bounds__(256) void gat_gemm_wmma(
    const float* __restrict__ X, const float* __restrict__ W,
    float* __restrict__ XW) {
  __shared__ __align__(16) bf16 sWt[FDIM][LDS_K];  // [n][k]  (W transposed)
  __shared__ __align__(16) bf16 sX[64][LDS_K];     // [m][k]

  const int tid = threadIdx.x;
  const int rowBase = blockIdx.x * 64;

  // stage W transposed: coalesced global reads (two rows of W per pass),
  // contiguous-in-k LDS stores
  for (int i = tid; i < 64 * FDIM; i += 256) {
    int rp = i >> 7;       // k-pair index 0..63
    int c  = i & 127;      // n
    sWt[c][2 * rp]     = (bf16)W[(size_t)(2 * rp) * FDIM + c];
    sWt[c][2 * rp + 1] = (bf16)W[(size_t)(2 * rp + 1) * FDIM + c];
  }
  // stage X tile 64x128 (guard the 32-row tail block)
  for (int i = tid; i < 64 * IN_CH; i += 256) {
    int r = i >> 7, c = i & 127;
    int gr = rowBase + r;
    sX[r][c] = (bf16)((gr < NNODES) ? X[(size_t)gr * IN_CH + c] : 0.0f);
  }
  __syncthreads();

  const int wave = tid >> 5;        // N-tile 0..7
  const int lane = tid & 31;
  const int lrow = lane & 15;       // M (for A) / N (for B,C)
  const int hi   = lane >> 4;       // lane-half select
  const int col  = wave * 16 + lrow;

  // B fragments: lane = N column; element i -> K = kb + hi*16 + i (contiguous)
  v16bf bfrag[4];
#pragma unroll
  for (int kk = 0; kk < 4; ++kk) {
    const int kb = kk * 32 + hi * 16;
    v8bf b0 = *(const v8bf*)&sWt[col][kb];
    v8bf b1 = *(const v8bf*)&sWt[col][kb + 8];
    bfrag[kk] = __builtin_shufflevector(b0, b1, 0, 1, 2, 3, 4, 5, 6, 7,
                                        8, 9, 10, 11, 12, 13, 14, 15);
  }

#pragma unroll
  for (int mt = 0; mt < 4; ++mt) {
    const int mrow = mt * 16 + lrow;
    if (rowBase + mt * 16 >= NNODES) break;   // whole-tile guard (N % 16 == 0)
    v8f acc = {};
#pragma unroll
    for (int kk = 0; kk < 4; ++kk) {
      const int kb = kk * 32;
      // A 16x32 16-bit layout: elem i -> K=kb+hi*16+i (contiguous, matches B)
      v8bf a0 = *(const v8bf*)&sX[mrow][kb + hi * 16];
      v8bf a1 = *(const v8bf*)&sX[mrow][kb + hi * 16 + 8];
      v16bf a = __builtin_shufflevector(a0, a1, 0, 1, 2, 3, 4, 5, 6, 7,
                                        8, 9, 10, 11, 12, 13, 14, 15);
      acc = __builtin_amdgcn_wmma_f32_16x16x32_bf16(
          /*neg_a=*/false, a, /*neg_b=*/false, bfrag[kk],
          /*c_mod=*/(short)0, acc, /*reuse_a=*/false, /*reuse_b=*/false);
    }
    // C layout: acc[r]: lanes0-15 -> M=r, lanes16-31 -> M=r+8; N=lrow
#pragma unroll
    for (int r = 0; r < 8; ++r) {
      int m = mt * 16 + r + hi * 8;
      XW[(size_t)(rowBase + m) * FDIM + col] = acc[r];
    }
  }
}

// ------------------------------------------------------------------
// 2) per-node attention logits: a_src[n,h], a_dst[n,h]
// ------------------------------------------------------------------
__global__ void gat_logits(const float* __restrict__ XW,
                           const float* __restrict__ att_src,
                           const float* __restrict__ att_dst,
                           float* __restrict__ a_src, float* __restrict__ a_dst) {
  int i = blockIdx.x * blockDim.x + threadIdx.x;  // node*HEADS + h
  if (i >= NNODES * HEADS) return;
  int n = i >> 2, h = i & 3;
  const float* row = XW + (size_t)n * FDIM + h * OUT_CH;
  const float* as  = att_src + h * OUT_CH;
  const float* ad  = att_dst + h * OUT_CH;
  float s = 0.f, d = 0.f;
#pragma unroll
  for (int c = 0; c < OUT_CH; ++c) {
    float v = row[c];
    s += v * as[c];
    d += v * ad[c];
  }
  a_src[i] = s;
  a_dst[i] = d;
}

// float atomic max via int/uint monotonic mapping (handles negatives)
__device__ __forceinline__ void atomicMaxF(float* addr, float value) {
  if (value >= 0.0f)
    atomicMax((int*)addr, __float_as_int(value));
  else
    atomicMin((unsigned int*)addr, __float_as_uint(value));
}

// ------------------------------------------------------------------
// 3) per-edge leaky-relu logits + segment max over dst
// ------------------------------------------------------------------
__global__ void gat_edge_max(const int* __restrict__ src, const int* __restrict__ dst,
                             const float* __restrict__ a_src, const float* __restrict__ a_dst,
                             float* __restrict__ alpha, float* __restrict__ amax) {
  long long e = (long long)blockIdx.x * blockDim.x + threadIdx.x;
  if (e >= EPRIME) return;
  int s, d;
  if (e < NEDGES) { s = src[e]; d = dst[e]; } else { s = d = (int)(e - NEDGES); }
  float4 vs = ((const float4*)a_src)[s];
  float4 vd = ((const float4*)a_dst)[d];
  float4 v;
  v.x = vs.x + vd.x; v.y = vs.y + vd.y; v.z = vs.z + vd.z; v.w = vs.w + vd.w;
  v.x = (v.x >= 0.f) ? v.x : NEG_SLOPE * v.x;
  v.y = (v.y >= 0.f) ? v.y : NEG_SLOPE * v.y;
  v.z = (v.z >= 0.f) ? v.z : NEG_SLOPE * v.z;
  v.w = (v.w >= 0.f) ? v.w : NEG_SLOPE * v.w;
  ((float4*)alpha)[e] = v;
  atomicMaxF(&amax[d * HEADS + 0], v.x);
  atomicMaxF(&amax[d * HEADS + 1], v.y);
  atomicMaxF(&amax[d * HEADS + 2], v.z);
  atomicMaxF(&amax[d * HEADS + 3], v.w);
}

// ------------------------------------------------------------------
// 4) exp(alpha - max) + segment sum over dst
// ------------------------------------------------------------------
__global__ void gat_edge_exp(const int* __restrict__ dst,
                             float* __restrict__ alpha,
                             const float* __restrict__ amax,
                             float* __restrict__ denom) {
  long long e = (long long)blockIdx.x * blockDim.x + threadIdx.x;
  if (e >= EPRIME) return;
  int d = (e < NEDGES) ? dst[e] : (int)(e - NEDGES);
  float4 v = ((const float4*)alpha)[e];
  float4 m = ((const float4*)amax)[d];
  float4 ex;
  ex.x = __expf(v.x - m.x);
  ex.y = __expf(v.y - m.y);
  ex.z = __expf(v.z - m.z);
  ex.w = __expf(v.w - m.w);
  ((float4*)alpha)[e] = ex;
  atomicAdd(&denom[d * HEADS + 0], ex.x);
  atomicAdd(&denom[d * HEADS + 1], ex.y);
  atomicAdd(&denom[d * HEADS + 2], ex.z);
  atomicAdd(&denom[d * HEADS + 3], ex.w);
}

// ------------------------------------------------------------------
// 5) weighted scatter-aggregate: out[dst] += xw[src] * alpha/denom
//    128 consecutive threads share one edge -> coalesced xw reads and
//    row-contiguous atomics into out.
// ------------------------------------------------------------------
__global__ void gat_scatter(const int* __restrict__ src, const int* __restrict__ dst,
                            const float* __restrict__ XW,
                            const float* __restrict__ alpha,
                            const float* __restrict__ denom,
                            float* __restrict__ out) {
  long long t = (long long)blockIdx.x * blockDim.x + threadIdx.x;
  if (t >= (long long)EPRIME * FDIM) return;
  long long e = t >> 7;
  int c = (int)(t & 127);
  int h = c >> 5;
  int s, d;
  if (e < NEDGES) { s = src[e]; d = dst[e]; } else { s = d = (int)(e - NEDGES); }
  float w = alpha[e * HEADS + h] / (denom[d * HEADS + h] + 1e-16f);
  atomicAdd(&out[(size_t)d * FDIM + c], XW[(size_t)s * FDIM + c] * w);
}

// ------------------------------------------------------------------
// 6) finalize: out = elu(out + bias)
// ------------------------------------------------------------------
__global__ void gat_finalize(float* __restrict__ out, const float* __restrict__ bias) {
  long long i = (long long)blockIdx.x * blockDim.x + threadIdx.x;
  if (i >= (long long)NNODES * FDIM) return;
  float v = out[i] + bias[i & 127];
  out[i] = (v > 0.f) ? v : (__expf(v) - 1.0f);
}

// ------------------------------------------------------------------
extern "C" void kernel_launch(void* const* d_in, const int* in_sizes, int n_in,
                              void* d_out, int out_size, void* d_ws, size_t ws_size,
                              hipStream_t stream) {
  const float* x       = (const float*)d_in[0];
  const int*   eidx    = (const int*)d_in[1];   // [2, E] flat: row0=src, row1=dst
  const float* W       = (const float*)d_in[2];
  const float* att_src = (const float*)d_in[3];
  const float* att_dst = (const float*)d_in[4];
  const float* bias    = (const float*)d_in[5];
  float*       out     = (float*)d_out;

  const int* src = eidx;
  const int* dst = eidx + NEDGES;

  // workspace carve-out (floats): ~85 MB total
  float* ws    = (float*)d_ws;
  float* xw    = ws;                              // N*128 = 12,800,000
  float* a_src = xw    + (size_t)NNODES * FDIM;
  float* a_dst = a_src + (size_t)NNODES * HEADS;
  float* amax  = a_dst + (size_t)NNODES * HEADS;
  float* denom = amax  + (size_t)NNODES * HEADS;
  float* alpha = denom + (size_t)NNODES * HEADS;  // EPRIME*4 = 6,800,000

  const int TB = 256;
  const long long nOut = (long long)NNODES * FDIM;

  // 0) init
  {
    int blocks = (int)((nOut + TB - 1) / TB);
    gat_init<<<blocks, TB, 0, stream>>>(out, amax, denom);
  }
  // 1) GEMM via WMMA: 64 rows per block
  gat_gemm_wmma<<<(NNODES + 63) / 64, TB, 0, stream>>>(x, W, xw);
  // 2) per-node logits
  {
    int total = NNODES * HEADS;
    gat_logits<<<(total + TB - 1) / TB, TB, 0, stream>>>(xw, att_src, att_dst, a_src, a_dst);
  }
  // 3) edge logits + segment max
  gat_edge_max<<<(EPRIME + TB - 1) / TB, TB, 0, stream>>>(src, dst, a_src, a_dst, alpha, amax);
  // 4) exp + segment sum
  gat_edge_exp<<<(EPRIME + TB - 1) / TB, TB, 0, stream>>>(dst, alpha, amax, denom);
  // 5) weighted scatter
  {
    long long total = (long long)EPRIME * FDIM;
    int blocks = (int)((total + TB - 1) / TB);
    gat_scatter<<<blocks, TB, 0, stream>>>(src, dst, xw, alpha, denom, out);
  }
  // 6) bias + ELU
  {
    int blocks = (int)((nOut + TB - 1) / TB);
    gat_finalize<<<blocks, TB, 0, stream>>>(out, bias);
  }
}
